// Transformer_Based_Model_13331578487358
// MI455X (gfx1250) — matrix operand order repaired and checked by reading, not compiled, not run on hardware
//
#include <hip/hip_runtime.h>
#include <hip/hip_bf16.h>
#include <math.h>

typedef __attribute__((ext_vector_type(16))) __bf16 v16bf;
typedef __attribute__((ext_vector_type(8)))  __bf16 v8bf;
typedef __attribute__((ext_vector_type(8)))  float  v8f;
typedef __attribute__((ext_vector_type(4)))  unsigned int v4u;
typedef __attribute__((ext_vector_type(8)))  int v8i;
typedef __attribute__((ext_vector_type(4)))  int v4i;

#define N_NODES   2048
#define B_BATCH   4
#define OUTF      512
#define INF_      512
#define ROWS      (B_BATCH * N_NODES)      // 8192
#define NEG_INF_F (-9.0e15f)
#define LAM_INIT  0.35550926757844716f     // 0.8 - 0.6*exp(-0.3)
#define ONE_MINUS_LAM_INIT 0.6444907324215528f

// TDM-staged B panel: 64 rows x K_CH bf16, padded 16B per 256B row -> 272B pitch
#define K_CH      128
#define B_ROWS    64
#define ROW_PITCH 272

// ---------------------------------------------------------------- scalars ----
__global__ __launch_bounds__(256) void k_scalars(
    const float* __restrict__ ll1, const float* __restrict__ lr1,
    const float* __restrict__ ll2, const float* __restrict__ lr2,
    const float* __restrict__ rel_table,
    const float* __restrict__ a_rel_pos, const float* __restrict__ a_rel_neg,
    float* __restrict__ sc)
{
    __shared__ float ra[256], rb[256];
    int t = threadIdx.x;
    ra[t] = ll1[t] * lr1[t];
    rb[t] = ll2[t] * lr2[t];
    __syncthreads();
    for (int s = 128; s > 0; s >>= 1) {
        if (t < s) { ra[t] += ra[t + s]; rb[t] += rb[t + s]; }
        __syncthreads();
    }
    if (t == 0) sc[0] = expf(ra[0]) - expf(rb[0]) + LAM_INIT;
    if (t < 6) {
        float sp = 0.f, sn = 0.f;
        for (int d = 0; d < 10; ++d) {
            float rv = rel_table[t * 10 + d];
            sp += rv * a_rel_pos[d];
            sn += rv * a_rel_neg[d];
        }
        sc[1 + t] = sp;
        sc[7 + t] = sn;
    }
}

// ------------------------------------------- f32 [R][C] -> bf16 [C][R] ------
__global__ __launch_bounds__(256) void k_transpose_f32_bf16(
    const float* __restrict__ in, __bf16* __restrict__ out, int R, int C)
{
    __shared__ float tile[32][33];
    size_t bofs = (size_t)blockIdx.z * (size_t)R * (size_t)C;
    const float* inb = in + bofs;
    __bf16* outb = out + bofs;
    int c0 = blockIdx.x * 32, r0 = blockIdx.y * 32;
    for (int r = threadIdx.y; r < 32; r += 8)
        tile[r][threadIdx.x] = inb[(size_t)(r0 + r) * C + c0 + threadIdx.x];
    __syncthreads();
    for (int r = threadIdx.y; r < 32; r += 8)
        outb[(size_t)(c0 + r) * R + r0 + threadIdx.x] = (__bf16)tile[threadIdx.x][r];
}

// ------------------------------------------------ TDM tile load (D# build) --
// Loads a 2D bf16 tile: 64 rows x K_CH cols from a [..][K] row-major matrix
// into LDS at lds_off, padding 16B after each 256B row (ROW_PITCH=272).
__device__ __forceinline__ void tdm_load_b_tile(const __bf16* gsrc,
                                                unsigned lds_off, int K)
{
    unsigned long long ga = (unsigned long long)(uintptr_t)gsrc;
    v4u g0;
    g0[0] = 1u;                                   // count=1 (valid), no gather
    g0[1] = lds_off;                              // LDS byte address
    g0[2] = (unsigned)(ga & 0xffffffffu);         // global_addr[31:0]
    g0[3] = (unsigned)((ga >> 32) & 0x01ffffffu)  // global_addr[56:32]
          | (2u << 30);                           // type=2 ("image")
    v8i g1;
    g1[0] = (int)((1u << 16)                      // data_size = 2 bytes
          | (1u << 20)                            // pad_enable
          | (5u << 22)                            // pad_interval: 64 DW = 256B
          | (3u << 25));                          // pad_amount: 4 DW = 16B
    g1[1] = (int)(((unsigned)K & 0xffffu) << 16); // tensor_dim0[15:0]
    g1[2] = (int)((((unsigned)K >> 16) & 0xffffu) // tensor_dim0[31:16]
          | ((unsigned)B_ROWS << 16));            // tensor_dim1[15:0] = 64
    g1[3] = (int)((unsigned)K_CH << 16);          // tensor_dim1 hi=0, tile_dim0=128
    g1[4] = (int)B_ROWS;                          // tile_dim1=64, tile_dim2=0
    g1[5] = (int)(unsigned)K;                     // tensor_dim0_stride[31:0]
    g1[6] = 0;                                    // stride0 hi, stride1 lo
    g1[7] = 0;                                    // stride1 hi
    v4i zz = {0, 0, 0, 0};
    v8i z8 = {0, 0, 0, 0, 0, 0, 0, 0};
    __builtin_amdgcn_tensor_load_to_lds(g0, g1, zz, zz, z8, 0);
}

// --------------------------------------------------------- bf16 WMMA GEMM ---
// C[M][Ncols] (f32) = A[M][K] (f32 -> bf16 on the fly) * Bt[Ncols][K]^T (bf16)
// block = 4 waves; each wave owns one 16x64 tile; B panel staged via TDM into
// double-buffered LDS; K stepped by 32 with v_wmma_f32_16x16x32_bf16.
__global__ __launch_bounds__(128) void k_gemm_bf16_tdm(
    const float* __restrict__ A, const __bf16* __restrict__ Bt,
    float* __restrict__ C, int K, int Ncols,
    size_t strideA, size_t strideB, size_t strideC)
{
    __shared__ __align__(16) char ldsB[2][B_ROWS * ROW_PITCH];

    const int tid  = threadIdx.x;
    const int wave = tid >> 5;
    const int lane = tid & 31;
    const int l15  = lane & 15;
    const int sel  = lane >> 4;                 // 0 or 1
    const int mTile = blockIdx.x * 4 + wave;    // 16 rows per wave
    const int nBase = blockIdx.y * 64;          // 64 cols per block
    const int z     = blockIdx.z;

    const float*  arow  = A  + (size_t)z * strideA + (size_t)(mTile * 16 + l15) * K;
    const __bf16* bbase = Bt + (size_t)z * strideB + (size_t)nBase * K;

    const int nch = K / K_CH;

    if (tid < 32) {   // wave 0 drives the Tensor Data Mover
        tdm_load_b_tile(bbase, (unsigned)(uintptr_t)(&ldsB[0][0]), K);
        __builtin_amdgcn_s_wait_tensorcnt(0);
    }
    __syncthreads();

    v8f acc0 = {}, acc1 = {}, acc2 = {}, acc3 = {};

    for (int c = 0; c < nch; ++c) {
        const int k0 = c * K_CH;
        if (tid < 32 && (c + 1) < nch)
            tdm_load_b_tile(bbase + (k0 + K_CH),
                            (unsigned)(uintptr_t)(&ldsB[(c + 1) & 1][0]), K);

        const char* bufp = ldsB[c & 1];
        __builtin_prefetch((const void*)(arow + k0 + K_CH), 0, 3);

#pragma unroll
        for (int kc = 0; kc < K_CH; kc += 32) {
            // A fragment: lane<16 -> K {k..k+7, k+16..k+23}; lane>=16 -> +8
            union { v16bf v; v8bf h[2]; } ua;
            ua.h[0] = __builtin_convertvector(*(const v8f*)(arow + k0 + kc + sel * 8), v8bf);
            ua.h[1] = __builtin_convertvector(*(const v8f*)(arow + k0 + kc + 16 + sel * 8), v8bf);

            // B fragments from LDS: lane holds col n = t*16+l15, K {k..k+7, k+16..k+23} (+8 for lane>=16)
            const char* bp = bufp + (size_t)l15 * ROW_PITCH + (size_t)(kc + sel * 8) * 2;
            union { v16bf v; v8bf h[2]; } ub0, ub1, ub2, ub3;
            ub0.h[0] = *(const v8bf*)(bp);
            ub0.h[1] = *(const v8bf*)(bp + 32);
            ub1.h[0] = *(const v8bf*)(bp + 16 * ROW_PITCH);
            ub1.h[1] = *(const v8bf*)(bp + 16 * ROW_PITCH + 32);
            ub2.h[0] = *(const v8bf*)(bp + 32 * ROW_PITCH);
            ub2.h[1] = *(const v8bf*)(bp + 32 * ROW_PITCH + 32);
            ub3.h[0] = *(const v8bf*)(bp + 48 * ROW_PITCH);
            ub3.h[1] = *(const v8bf*)(bp + 48 * ROW_PITCH + 32);

            acc0 = __builtin_amdgcn_wmma_f32_16x16x32_bf16(false, ua.v, false, ub0.v, (short)0, acc0, false, false);
            acc1 = __builtin_amdgcn_wmma_f32_16x16x32_bf16(false, ua.v, false, ub1.v, (short)0, acc1, false, false);
            acc2 = __builtin_amdgcn_wmma_f32_16x16x32_bf16(false, ua.v, false, ub2.v, (short)0, acc2, false, false);
            acc3 = __builtin_amdgcn_wmma_f32_16x16x32_bf16(false, ua.v, false, ub3.v, (short)0, acc3, false, false);
        }

        if (tid < 32 && (c + 1) < nch)
            __builtin_amdgcn_s_wait_tensorcnt(0);
        __syncthreads();   // publish next buffer / retire reads of current
    }

    // C/D layout: lane -> col n=l15; VGPR r -> row sel*8 + r
    float* crow = C + (size_t)z * strideC + (size_t)(mTile * 16 + sel * 8) * Ncols + nBase + l15;
#pragma unroll
    for (int r = 0; r < 8; ++r) {
        crow[(size_t)r * Ncols + 0]  = acc0[r];
        crow[(size_t)r * Ncols + 16] = acc1[r];
        crow[(size_t)r * Ncols + 32] = acc2[r];
        crow[(size_t)r * Ncols + 48] = acc3[r];
    }
}

// ----------------------------------------------- per-row attention dots -----
__global__ __launch_bounds__(256) void k_rowdots(
    const float* __restrict__ Wh,
    const float* __restrict__ alp, const float* __restrict__ arp,
    const float* __restrict__ aln, const float* __restrict__ arn,
    float* __restrict__ vecs)
{
    int wave = threadIdx.x >> 5, lane = threadIdx.x & 31;
    int row = blockIdx.x * 8 + wave;
    const float* w = Wh + (size_t)row * OUTF;
    float s0 = 0.f, s1 = 0.f, s2 = 0.f, s3 = 0.f;
#pragma unroll
    for (int t = 0; t < 8; ++t) {
        int idx = lane + t * 32;             // 0..255
        float wp = w[idx], wn = w[256 + idx];
        s0 += wp * alp[idx];
        s1 += wp * arp[idx];
        s2 += wn * aln[idx];
        s3 += wn * arn[idx];
    }
#pragma unroll
    for (int off = 16; off > 0; off >>= 1) {
        s0 += __shfl_xor(s0, off, 32);
        s1 += __shfl_xor(s1, off, 32);
        s2 += __shfl_xor(s2, off, 32);
        s3 += __shfl_xor(s3, off, 32);
    }
    if (lane == 0) {
        vecs[row]             = s0;   // left_pos
        vecs[ROWS + row]      = s1;   // right_pos
        vecs[2 * ROWS + row]  = s2;   // left_neg
        vecs[3 * ROWS + row]  = s3;   // right_neg
    }
}

// -------------------------------------------- masked dual softmax rows ------
__global__ __launch_bounds__(256) void k_attention(
    const int* __restrict__ adj, const float* __restrict__ vecs,
    const float* __restrict__ sc, float* __restrict__ att)
{
    const int i = blockIdx.x, b = blockIdx.y, tid = threadIdx.x;
    const int* arow = adj + ((size_t)b * N_NODES + i) * N_NODES;

    __shared__ float srp[6], srn[6];
    __shared__ float redA[256], redB[256];
    if (tid < 6) { srp[tid] = sc[1 + tid]; srn[tid] = sc[7 + tid]; }
    __syncthreads();

    const float lam = sc[0];
    const float lp  = vecs[(size_t)b * N_NODES + i];
    const float lnn = vecs[2 * ROWS + (size_t)b * N_NODES + i];

    float ep[8], en[8];
    float mp = NEG_INF_F, mn = NEG_INF_F;
#pragma unroll
    for (int t = 0; t < 8; ++t) {
        int j = t * 256 + tid;
        int a = arow[j];
        float vp = lp  + vecs[ROWS + (size_t)b * N_NODES + j]     + srp[a];
        float vn = lnn + vecs[3 * ROWS + (size_t)b * N_NODES + j] + srn[a];
        vp = (vp >= 0.f) ? vp : 0.2f * vp;
        vn = (vn >= 0.f) ? vn : 0.2f * vn;
        bool m = (a > 0);
        ep[t] = m ? vp : NEG_INF_F;
        en[t] = m ? vn : NEG_INF_F;
        mp = fmaxf(mp, ep[t]);
        mn = fmaxf(mn, en[t]);
    }
    redA[tid] = mp; redB[tid] = mn; __syncthreads();
    for (int s = 128; s > 0; s >>= 1) {
        if (tid < s) {
            redA[tid] = fmaxf(redA[tid], redA[tid + s]);
            redB[tid] = fmaxf(redB[tid], redB[tid + s]);
        }
        __syncthreads();
    }
    mp = redA[0]; mn = redB[0];
    __syncthreads();

    float sp = 0.f, sn = 0.f;
#pragma unroll
    for (int t = 0; t < 8; ++t) {
        ep[t] = __expf(ep[t] - mp);
        en[t] = __expf(en[t] - mn);
        sp += ep[t]; sn += en[t];
    }
    redA[tid] = sp; redB[tid] = sn; __syncthreads();
    for (int s = 128; s > 0; s >>= 1) {
        if (tid < s) { redA[tid] += redA[tid + s]; redB[tid] += redB[tid + s]; }
        __syncthreads();
    }
    const float rsp = 1.f / redA[0], rsn = 1.f / redB[0];

    float* orow = att + ((size_t)b * N_NODES + i) * N_NODES;
#pragma unroll
    for (int t = 0; t < 8; ++t) {
        int j = t * 256 + tid;
        orow[j] = ep[t] * rsp - lam * en[t] * rsn;
    }
}

// ---------------------------------------------------- layernorm + gelu ------
__global__ __launch_bounds__(256) void k_lngelu(
    const float* __restrict__ hp, const float* __restrict__ gamma,
    const float* __restrict__ beta, float* __restrict__ out)
{
    const int row = blockIdx.x, tid = threadIdx.x;
    const float* x = hp + (size_t)row * OUTF;
    __shared__ float red[256];
    float v0 = x[tid], v1 = x[tid + 256];

    red[tid] = v0 + v1; __syncthreads();
    for (int s = 128; s > 0; s >>= 1) { if (tid < s) red[tid] += red[tid + s]; __syncthreads(); }
    float mu = red[0] * (1.f / 512.f);
    __syncthreads();

    float d0 = v0 - mu, d1 = v1 - mu;
    red[tid] = d0 * d0 + d1 * d1; __syncthreads();
    for (int s = 128; s > 0; s >>= 1) { if (tid < s) red[tid] += red[tid + s]; __syncthreads(); }
    float inv = rsqrtf(red[0] * (1.f / 512.f) + 1e-5f);

    float* orow = out + (size_t)row * OUTF;
#pragma unroll
    for (int e = 0; e < 2; ++e) {
        int o = tid + e * 256;
        float d = (e == 0) ? d0 : d1;
        float y = (d * inv * gamma[o] + beta[o]) * ONE_MINUS_LAM_INIT;
        orow[o] = 0.5f * y * (1.f + erff(y * 0.70710678118f));
    }
}

// ------------------------------------------------------------- launcher -----
extern "C" void kernel_launch(void* const* d_in, const int* in_sizes, int n_in,
                              void* d_out, int out_size, void* d_ws, size_t ws_size,
                              hipStream_t stream)
{
    const float* h          = (const float*)d_in[0];
    const int*   adj        = (const int*)  d_in[1];
    const float* W          = (const float*)d_in[2];
    const float* a_left_pos = (const float*)d_in[3];
    const float* a_right_pos= (const float*)d_in[4];
    const float* a_left_neg = (const float*)d_in[5];
    const float* a_right_neg= (const float*)d_in[6];
    const float* rel_table  = (const float*)d_in[7];
    const float* a_rel_pos  = (const float*)d_in[8];
    const float* a_rel_neg  = (const float*)d_in[9];
    const float* ll1        = (const float*)d_in[10];
    const float* lr1        = (const float*)d_in[11];
    const float* ll2        = (const float*)d_in[12];
    const float* lr2        = (const float*)d_in[13];
    const float* ln_gamma   = (const float*)d_in[14];
    const float* ln_beta    = (const float*)d_in[15];

    float* out_gelu = (float*)d_out;                                   // [4][2048][512]
    float* out_att  = (float*)d_out + (size_t)ROWS * OUTF;             // [4][2048][2048]

    char* ws = (char*)d_ws;
    __bf16* Wt   = (__bf16*)(ws);                                      // [512][512] bf16
    __bf16* WhT  = (__bf16*)(ws + 524288);                             // [4][512][2048] bf16
    float*  Wh   = (float*)(ws + 524288 + 8388608);                    // [8192][512] f32 (reused as h_prime)
    float*  vecs = (float*)(ws + 524288 + 8388608 + 16777216);         // [4][8192]
    float*  sc   = (float*)(ws + 524288 + 8388608 + 16777216 + 131072);// scalars

    // 1) lambda + 6+6 relation scalars
    k_scalars<<<1, 256, 0, stream>>>(ll1, lr1, ll2, lr2, rel_table, a_rel_pos, a_rel_neg, sc);

    // 2) Wt[o][i] = bf16(W[i][o])   (R=512 rows, C=512 cols)
    k_transpose_f32_bf16<<<dim3(16, 16, 1), dim3(32, 8), 0, stream>>>(W, Wt, 512, 512);

    // 3) Wh = h @ W  (8192x512x512, TDM-staged bf16 WMMA, f32 accum)
    k_gemm_bf16_tdm<<<dim3(ROWS / 64, OUTF / 64, 1), 128, 0, stream>>>(
        h, Wt, Wh, INF_, OUTF, 0, 0, 0);

    // 4) per-row attention dot products
    k_rowdots<<<ROWS / 8, 256, 0, stream>>>(Wh, a_left_pos, a_right_pos, a_left_neg, a_right_neg, vecs);

    // 5) WhT[b][o][j] = bf16(Wh[b][j][o])  (R=2048, C=512 per batch)
    k_transpose_f32_bf16<<<dim3(16, 64, B_BATCH), dim3(32, 8), 0, stream>>>(Wh, WhT, 2048, 512);

    // 6) attention rows -> out_att (also GEMM2's A matrix)
    k_attention<<<dim3(N_NODES, B_BATCH), 256, 0, stream>>>(adj, vecs, sc, out_att);

    // 7) h_prime[b] = attention[b] @ Wh[b]  (2048x512x2048 per batch, bf16 WMMA)
    //    h_prime overwrites the Wh f32 buffer (its consumers finished in 4/5)
    k_gemm_bf16_tdm<<<dim3(N_NODES / 64, OUTF / 64, B_BATCH), 128, 0, stream>>>(
        out_att, WhT, Wh, N_NODES, OUTF,
        (size_t)N_NODES * N_NODES, (size_t)OUTF * N_NODES, (size_t)N_NODES * OUTF);

    // 8) layernorm + scale + exact gelu
    k_lngelu<<<ROWS, 256, 0, stream>>>(Wh, ln_gamma, ln_beta, out_gelu);
}
